// SAMRelations_83391085019189
// MI455X (gfx1250) — matrix operand order repaired and checked by reading, not compiled, not run on hardware
//
#include <hip/hip_runtime.h>

typedef __bf16 bf16_t;
typedef bf16_t v16bf __attribute__((ext_vector_type(16)));
typedef float v8f __attribute__((ext_vector_type(8)));

#define HW   16384
#define CDIM 1280
#define ODIM 512

__device__ __forceinline__ unsigned short f32_to_bf16_rne(float f) {
  union { float f; unsigned int u; } x; x.f = f;
  unsigned int r = (x.u + 0x7FFFu + ((x.u >> 16) & 1u)) >> 16;
  return (unsigned short)r;
}

__global__ __launch_bounds__(256)
void convert_bf16_kernel(const float* __restrict__ src, unsigned short* __restrict__ dst, long n) {
  long idx = (long)blockIdx.x * 256 + threadIdx.x;
  if (idx < n) dst[idx] = f32_to_bf16_rne(src[idx]);
}

__global__ __launch_bounds__(256)
void copy_f32_kernel(const float* __restrict__ src, float* __restrict__ dst, long n) {
  long idx = (long)blockIdx.x * 256 + threadIdx.x;
  if (idx < n) dst[idx] = src[idx];
}

// C[m, n] = sum_k A[m, k] * B[n, k]   (A: M x K bf16 row-major, B: N x K bf16 row-major)
// Block tile 128x128, BK=32, double-buffered LDS; 8 waves, each wave owns 32x64.
// fuse==1: C[off] = relu(acc) + resid[off]
__global__ __launch_bounds__(256)
void gemm_bf16_wmma_kernel(const unsigned short* __restrict__ A,
                           const unsigned short* __restrict__ B,
                           float* __restrict__ C,
                           const float* __restrict__ resid,
                           int N, int K, int lda, int ldb, int fuse)
{
  __shared__ unsigned short ldsA[2][128 * 32];
  __shared__ unsigned short ldsB[2][128 * 32];

  const int t      = threadIdx.x;
  const int lane   = t & 31;
  const int wid    = t >> 5;
  const int wave_m = wid & 3;   // 0..3 -> M offset *32
  const int wave_n = wid >> 2;  // 0..1 -> N offset *64
  const int half   = lane >> 4;
  const int mn     = lane & 15;

  const long blockM = (long)blockIdx.y * 128;
  const long blockN = (long)blockIdx.x * 128;

  v8f acc[2][4] = {};

  // staging: each thread copies 32B of A-tile and 32B of B-tile per K-step
  const int arow = t >> 1, achunk = t & 1;   // 128 rows x 2 chunks of 16 bf16

  const unsigned short* gA = A + (blockM + arow) * (long)lda + achunk * 16;
  const unsigned short* gB = B + (blockN + arow) * (long)ldb + achunk * 16;

  uint4 ra0, ra1, rb0, rb1;

  // preload k-tile 0
  {
    const uint4* pa = (const uint4*)(gA);
    ra0 = pa[0]; ra1 = pa[1];
    const uint4* pb = (const uint4*)(gB);
    rb0 = pb[0]; rb1 = pb[1];
    uint4* da = (uint4*)(&ldsA[0][arow * 32 + achunk * 16]);
    da[0] = ra0; da[1] = ra1;
    uint4* db = (uint4*)(&ldsB[0][arow * 32 + achunk * 16]);
    db[0] = rb0; db[1] = rb1;
  }
  __syncthreads();

  const int nk = K >> 5;
  for (int ki = 0; ki < nk; ++ki) {
    const int cur = ki & 1;
    const int nxt = cur ^ 1;
    const bool more = (ki + 1) < nk;

    // issue next tile's global loads early (latency hides under WMMA below)
    if (more) {
      const int k1 = (ki + 1) << 5;
      const uint4* pa = (const uint4*)(gA + k1);
      ra0 = pa[0]; ra1 = pa[1];
      const uint4* pb = (const uint4*)(gB + k1);
      rb0 = pb[0]; rb1 = pb[1];
      if (ki + 2 < nk) {
        __builtin_prefetch(gA + ((ki + 2) << 5), 0, 1);
        __builtin_prefetch(gB + ((ki + 2) << 5), 0, 1);
      }
    }

    union Frag { unsigned int u[8]; v16bf v; };
    Frag fa[2], fb[4];

    // A fragment: 16-bit A 16x32 layout; lane(0-15)=row, packed pair v holds
    // K = (v<4 ? 2v : 2v+8) + 8*(lane/16)
#pragma unroll
    for (int mi = 0; mi < 2; ++mi) {
      const int Ma = wave_m * 32 + mi * 16;
#pragma unroll
      for (int v = 0; v < 8; ++v) {
        const int kk = ((v < 4) ? (2 * v) : (2 * v + 8)) + 8 * half;
        fa[mi].u[v] = *(const unsigned int*)(&ldsA[cur][(Ma + mn) * 32 + kk]);
      }
    }
    // B fragment: 32x16; column lane%16, packed pair v holds K = (v<4 ? 2v : 2v+8) + 8*(lane/16)
#pragma unroll
    for (int ni = 0; ni < 4; ++ni) {
      const int Na = wave_n * 64 + ni * 16;
#pragma unroll
      for (int v = 0; v < 8; ++v) {
        const int kk = ((v < 4) ? (2 * v) : (2 * v + 8)) + 8 * half;
        fb[ni].u[v] = *(const unsigned int*)(&ldsB[cur][(Na + mn) * 32 + kk]);
      }
    }

#pragma unroll
    for (int mi = 0; mi < 2; ++mi)
#pragma unroll
      for (int ni = 0; ni < 4; ++ni)
        acc[mi][ni] = __builtin_amdgcn_wmma_f32_16x16x32_bf16(
            false, fa[mi].v, false, fb[ni].v, (short)0, acc[mi][ni], false, false);

    // write next tile into the other LDS buffer (loadcnt wait lands here, after WMMAs)
    if (more) {
      uint4* da = (uint4*)(&ldsA[nxt][arow * 32 + achunk * 16]);
      da[0] = ra0; da[1] = ra1;
      uint4* db = (uint4*)(&ldsB[nxt][arow * 32 + achunk * 16]);
      db[0] = rb0; db[1] = rb1;
    }
    __syncthreads();
  }

  // Epilogue: C/D layout: VGPR r -> M = r + 8*(lane/16), N = lane%16
#pragma unroll
  for (int mi = 0; mi < 2; ++mi) {
#pragma unroll
    for (int ni = 0; ni < 4; ++ni) {
      const long gn = blockN + wave_n * 64 + ni * 16 + mn;
#pragma unroll
      for (int r = 0; r < 8; ++r) {
        const long gm = blockM + wave_m * 32 + mi * 16 + r + 8 * half;
        float val = acc[mi][ni][r];
        const long off = gm * (long)N + gn;
        if (fuse) val = (val > 0.f ? val : 0.f) + resid[off];
        C[off] = val;
      }
    }
  }
}

// Xm[o2*HW + n] = sum_o v2[o2,o] * Y1[n,o]
__global__ __launch_bounds__(256)
void xm_kernel(const float* __restrict__ Y1, const float* __restrict__ v2,
               float* __restrict__ xm) {
  int idx = blockIdx.x * 256 + threadIdx.x;
  if (idx >= HW * 19) return;
  int n = idx / 19, o2 = idx % 19;
  const float* yrow = Y1 + (long)n * ODIM;
  const float* vrow = v2 + o2 * ODIM;
  float s = 0.f;
  for (int o = 0; o < ODIM; ++o) s += vrow[o] * yrow[o];
  xm[(long)o2 * HW + n] = s;
}

// per-n: aw = x_iv @ intra_m (19x1024); softmax; icab = sf @ intra_m^T; cab = icab @ inter_m
__global__ __launch_bounds__(128)
void chain_softmax_kernel(const float* __restrict__ xm, const float* __restrict__ intra,
                          const float* __restrict__ inter, const int* __restrict__ i_ptr,
                          float* __restrict__ cab) {
  __shared__ float s_xiv[19];
  __shared__ float s_red[128];
  __shared__ float s_acc[19];
  const int n = blockIdx.x, t = threadIdx.x;
  const int ii = *i_ptr;
  const float* intra_m = intra + (long)ii * 19 * 1024;
  const float* inter_m = inter + (long)ii * 19 * 19;

  if (t < 19) { s_xiv[t] = xm[(long)n * 19 + t]; s_acc[t] = 0.f; }
  __syncthreads();

  float a[8];
  float mx = -3.402823466e38f;
#pragma unroll
  for (int j = 0; j < 8; ++j) {
    const int k = t * 8 + j;
    float s = 0.f;
#pragma unroll
    for (int c = 0; c < 19; ++c) s += s_xiv[c] * intra_m[c * 1024 + k];
    a[j] = s;
    mx = fmaxf(mx, s);
  }
  s_red[t] = mx; __syncthreads();
  for (int off = 64; off > 0; off >>= 1) {
    if (t < off) s_red[t] = fmaxf(s_red[t], s_red[t + off]);
    __syncthreads();
  }
  mx = s_red[0]; __syncthreads();

  float sum = 0.f;
#pragma unroll
  for (int j = 0; j < 8; ++j) { a[j] = expf(a[j] - mx); sum += a[j]; }
  s_red[t] = sum; __syncthreads();
  for (int off = 64; off > 0; off >>= 1) {
    if (t < off) s_red[t] += s_red[t + off];
    __syncthreads();
  }
  const float inv = 1.f / s_red[0];

  float accl[19];
#pragma unroll
  for (int c = 0; c < 19; ++c) accl[c] = 0.f;
#pragma unroll
  for (int j = 0; j < 8; ++j) {
    const int k = t * 8 + j;
    const float p = a[j] * inv;
#pragma unroll
    for (int c = 0; c < 19; ++c) accl[c] += p * intra_m[c * 1024 + k];
  }
  for (int c = 0; c < 19; ++c) atomicAdd(&s_acc[c], accl[c]);
  __syncthreads();

  if (t < 19) {
    float s = 0.f;
#pragma unroll
    for (int c = 0; c < 19; ++c) s += s_acc[c] * inter_m[c * 19 + t];
    cab[(long)n * 19 + t] = s;
  }
}

// h2bf[m*512 + o] = bf16( sum_cc w1[o,cc] * cab[cc*HW + m] )
__global__ __launch_bounds__(256)
void h2_kernel(const float* __restrict__ cab, const float* __restrict__ w1,
               unsigned short* __restrict__ h2) {
  long idx = (long)blockIdx.x * 256 + threadIdx.x;
  if (idx >= (long)HW * ODIM) return;
  const int m = (int)(idx >> 9);
  const int o = (int)(idx & 511);
  float s = 0.f;
#pragma unroll
  for (int cc = 0; cc < 19; ++cc) s += w1[o * 19 + cc] * cab[(long)cc * HW + m];
  h2[(long)m * ODIM + o] = f32_to_bf16_rne(s);
}

extern "C" void kernel_launch(void* const* d_in, const int* in_sizes, int n_in,
                              void* d_out, int out_size, void* d_ws, size_t ws_size,
                              hipStream_t stream) {
  const float* x     = (const float*)d_in[0];
  const float* v1    = (const float*)d_in[1];
  const float* v2    = (const float*)d_in[2];
  const float* w1    = (const float*)d_in[3];
  const float* w2    = (const float*)d_in[4];
  const float* intra = (const float*)d_in[5];
  const float* inter = (const float*)d_in[6];
  const int*   i_ptr = (const int*)d_in[7];
  float* out = (float*)d_out;

  char* ws = (char*)d_ws;
  unsigned short* Xbf  = (unsigned short*)ws;                       // 16384*1280
  unsigned short* v1bf = Xbf + (long)HW * CDIM;                     // 512*1280
  unsigned short* w2bf = v1bf + 512L * 1280;                        // 1280*512
  float* Y1   = (float*)(w2bf + 1280L * 512);                       // 16384*512
  float* Xm   = Y1 + (long)HW * ODIM;                               // 19*16384
  float* cabf = Xm + 19L * HW;                                      // 16384*19
  unsigned short* h2bf = (unsigned short*)(cabf + (long)HW * 19);   // 16384*512

  const long nX = (long)HW * CDIM;
  convert_bf16_kernel<<<(nX + 255) / 256, 256, 0, stream>>>(x + 128L * CDIM, Xbf, nX);
  convert_bf16_kernel<<<(512L * 1280 + 255) / 256, 256, 0, stream>>>(v1, v1bf, 512L * 1280);
  convert_bf16_kernel<<<(1280L * 512 + 255) / 256, 256, 0, stream>>>(w2, w2bf, 1280L * 512);

  // GEMM1: Y1[n,o] = sum_c Xbf[n,c] * v1bf[o,c]   (M=16384, N=512, K=1280)
  dim3 g1(512 / 128, HW / 128);
  gemm_bf16_wmma_kernel<<<g1, 256, 0, stream>>>(Xbf, v1bf, Y1, nullptr, 512, 1280, 1280, 1280, 0);

  xm_kernel<<<(HW * 19 + 255) / 256, 256, 0, stream>>>(Y1, v2, Xm);
  chain_softmax_kernel<<<HW, 128, 0, stream>>>(Xm, intra, inter, i_ptr, cabf);
  h2_kernel<<<(int)(((long)HW * ODIM + 255) / 256), 256, 0, stream>>>(cabf, w1, h2bf);

  // GEMM2: out[(m+128)*1280 + c] = relu(sum_o h2bf[m,o]*w2bf[c,o]) + x[(m+128)*1280 + c]
  dim3 g2(CDIM / 128, HW / 128);
  gemm_bf16_wmma_kernel<<<g2, 256, 0, stream>>>(h2bf, w2bf, out + 128L * CDIM,
                                                x + 128L * CDIM, CDIM, 512, 512, 512, 1);

  // cls token row
  copy_f32_kernel<<<(128L * CDIM + 255) / 256, 256, 0, stream>>>(x, out, 128L * CDIM);
}
